// LearnableSimilarityAttention_69303592288274
// MI455X (gfx1250) — matrix operand order repaired and checked by reading, not compiled, not run on hardware
//
#include <hip/hip_runtime.h>
#include <hip/hip_bf16.h>

// Problem constants (match reference)
#define Bz 4
#define SQ 512
#define SK 512
#define Dd 512
#define Ss 128

typedef __attribute__((ext_vector_type(16))) __bf16 v16bf;
typedef __attribute__((ext_vector_type(8)))  __bf16 v8bf;
typedef __attribute__((ext_vector_type(8)))  float  v8f;

// ---------------------------------------------------------------------------
// f32 -> bf16 elementwise convert
// ---------------------------------------------------------------------------
__global__ void cvt_f32_bf16(const float* __restrict__ src, __bf16* __restrict__ dst, long n) {
    long i = (long)blockIdx.x * blockDim.x + threadIdx.x;
    long stride = (long)gridDim.x * blockDim.x;
    for (; i < n; i += stride) dst[i] = (__bf16)src[i];
}

// ---------------------------------------------------------------------------
// batched transpose + convert: dst[b][c][r] = (bf16) src[b][r][c]
// ---------------------------------------------------------------------------
__global__ void transpose_cvt(const float* __restrict__ src, __bf16* __restrict__ dst,
                              int rows, int cols) {
    int b = blockIdx.y;
    const float* s = src + (long)b * rows * cols;
    __bf16*      d = dst + (long)b * rows * cols;
    long n = (long)rows * cols;
    long i = (long)blockIdx.x * blockDim.x + threadIdx.x;
    long stride = (long)gridDim.x * blockDim.x;
    for (; i < n; i += stride) {
        long r = i / cols, c = i % cols;
        d[c * (long)rows + r] = (__bf16)s[i];
    }
}

// ---------------------------------------------------------------------------
// bf16 WMMA GEMM with N-direction register blocking:
//   C[b][M][N] = A[b][M][K] * BT[b][N][K]^T (+ bias[N])
// One wave computes a 16x64 tile: one A fragment is reused against four B
// fragments per 32-wide K step (4 independent v_wmma_f32_16x16x32_bf16,
// no D->A/B hazard).  BT is pre-transposed so every fragment load is a
// contiguous global_load_b128.
// ---------------------------------------------------------------------------
__global__ __launch_bounds__(32)
void gemm_bf16_wmma(const __bf16* __restrict__ A, const __bf16* __restrict__ BT,
                    const float* __restrict__ bias,
                    float* __restrict__ Cf, __bf16* __restrict__ Cb,
                    int M, int N, int K,
                    long strideA, long strideB, long strideC) {
    const int m0 = blockIdx.x * 16;
    const int n0 = blockIdx.y * 64;
    const int b  = blockIdx.z;
    const int l  = threadIdx.x & 31;

    const __bf16* Ab = A  + (long)b * strideA;
    const __bf16* Bb = BT + (long)b * strideB;

    const int row = m0 + (l & 15);           // A-fragment row (M)
    const int kbA = (l >> 4) * 16;           // A: 16 contiguous K per lane half
    const int kbB = (l >> 4) * 16;           // B: 16 contiguous K per lane half

    const __bf16* arow = Ab + (long)row * K;
    const __bf16* brow0 = Bb + (long)(n0      + (l & 15)) * K;
    const __bf16* brow1 = Bb + (long)(n0 + 16 + (l & 15)) * K;
    const __bf16* brow2 = Bb + (long)(n0 + 32 + (l & 15)) * K;
    const __bf16* brow3 = Bb + (long)(n0 + 48 + (l & 15)) * K;

    v8f c0 = {}, c1 = {}, c2 = {}, c3 = {};
    for (int k0 = 0; k0 < K; k0 += 32) {
        v8bf alo = *(const v8bf*)(arow + k0 + kbA);
        v8bf ahi = *(const v8bf*)(arow + k0 + 8 + kbA);
        v16bf a  = __builtin_shufflevector(alo, ahi,
                     0,1,2,3,4,5,6,7,8,9,10,11,12,13,14,15);
        v16bf b0 = *(const v16bf*)(brow0 + k0 + kbB);
        v16bf b1 = *(const v16bf*)(brow1 + k0 + kbB);
        v16bf b2 = *(const v16bf*)(brow2 + k0 + kbB);
        v16bf b3 = *(const v16bf*)(brow3 + k0 + kbB);
        c0 = __builtin_amdgcn_wmma_f32_16x16x32_bf16(false, a, false, b0, (short)0, c0, false, false);
        c1 = __builtin_amdgcn_wmma_f32_16x16x32_bf16(false, a, false, b1, (short)0, c1, false, false);
        c2 = __builtin_amdgcn_wmma_f32_16x16x32_bf16(false, a, false, b2, (short)0, c2, false, false);
        c3 = __builtin_amdgcn_wmma_f32_16x16x32_bf16(false, a, false, b3, (short)0, c3, false, false);
    }

    // D layout: VGPR r -> M = m0 + (l>>4)*8 + r, N = n_tile + (l&15)
    const int mbase = m0 + (l >> 4) * 8;
    v8f acc[4] = {c0, c1, c2, c3};
    #pragma unroll
    for (int j = 0; j < 4; ++j) {
        const int col = n0 + j * 16 + (l & 15);
        const float bv = bias ? bias[col] : 0.0f;
        #pragma unroll
        for (int r = 0; r < 8; ++r) {
            float v = acc[j][r] + bv;
            long idx = (long)b * strideC + (long)(mbase + r) * N + col;
            if (Cf) Cf[idx] = v;
            if (Cb) Cb[idx] = (__bf16)v;
        }
    }
}

// ---------------------------------------------------------------------------
// Fused scores + mask + softmax.
// One block = one (b,q).  scores[k] = sum_s relu(qh[q,s]+kh[k,s]+b1[s])*w2[s] + b2
// Fusing removes the 537MB 'hidden' tensor of the reference entirely.
// ---------------------------------------------------------------------------
__global__ __launch_bounds__(256)
void scores_softmax(const float* __restrict__ qh, const float* __restrict__ kh,
                    const float* __restrict__ b1, const float* __restrict__ W2,
                    const float* __restrict__ b2, const int* __restrict__ mask,
                    float* __restrict__ attn_f32, __bf16* __restrict__ attn_bf16) {
    const int q = blockIdx.x;
    const int b = blockIdx.y;
    const int t = threadIdx.x;

    __shared__ __align__(16) float qbuf[Ss];   // qh row + b1 (bias folded once)
    __shared__ __align__(16) float wv[Ss];     // w2
    __shared__ __align__(16) float sc[SK];
    __shared__ float part[256];

    if (t < Ss) {
        qbuf[t] = qh[((long)(b * SQ + q)) * Ss + t] + b1[t];
        wv[t]   = W2[t];
    }
    __syncthreads();

    const float b2v = b2[0];
    const long  mrow = ((long)(b * SQ + q)) * SK;

    for (int k = t; k < SK; k += 256) {
        const float4* kr = (const float4*)(kh + ((long)(b * SK + k)) * Ss);
        // hide L2 latency of the next row this thread will process
        if (k + 256 < SK)
            __builtin_prefetch(kh + ((long)(b * SK + k + 256)) * Ss, 0, 1);
        const float4* qr = (const float4*)qbuf;
        const float4* wr = (const float4*)wv;
        float acc = 0.0f;
        #pragma unroll 8
        for (int s = 0; s < Ss / 4; ++s) {
            float4 kv = kr[s]; float4 qv = qr[s]; float4 w = wr[s];
            acc = fmaf(fmaxf(qv.x + kv.x, 0.0f), w.x, acc);
            acc = fmaf(fmaxf(qv.y + kv.y, 0.0f), w.y, acc);
            acc = fmaf(fmaxf(qv.z + kv.z, 0.0f), w.z, acc);
            acc = fmaf(fmaxf(qv.w + kv.w, 0.0f), w.w, acc);
        }
        float scv = acc + b2v;
        if (mask[mrow + k] == 0) scv = -1e9f;
        sc[k] = scv;
    }
    __syncthreads();

    // block max
    float lm = -INFINITY;
    for (int k = t; k < SK; k += 256) lm = fmaxf(lm, sc[k]);
    part[t] = lm; __syncthreads();
    for (int ofs = 128; ofs > 0; ofs >>= 1) {
        if (t < ofs) part[t] = fmaxf(part[t], part[t + ofs]);
        __syncthreads();
    }
    const float mx = part[0];
    __syncthreads();

    // exp + block sum
    float ls = 0.0f;
    for (int k = t; k < SK; k += 256) {
        float e = __expf(sc[k] - mx);
        sc[k] = e;
        ls += e;
    }
    part[t] = ls; __syncthreads();
    for (int ofs = 128; ofs > 0; ofs >>= 1) {
        if (t < ofs) part[t] += part[t + ofs];
        __syncthreads();
    }
    const float inv = 1.0f / part[0];

    for (int k = t; k < SK; k += 256) {
        float a = sc[k] * inv;
        attn_f32[mrow + k]  = a;
        attn_bf16[mrow + k] = (__bf16)a;
    }
}

// ---------------------------------------------------------------------------
// Host-side launch
// ---------------------------------------------------------------------------
extern "C" void kernel_launch(void* const* d_in, const int* in_sizes, int n_in,
                              void* d_out, int out_size, void* d_ws, size_t ws_size,
                              hipStream_t stream) {
    (void)in_sizes; (void)n_in; (void)out_size; (void)ws_size;

    const float* query = (const float*)d_in[0];   // [B,SQ,D]
    const float* key   = (const float*)d_in[1];   // [B,SK,D]
    const float* value = (const float*)d_in[2];   // [B,SK,D]
    const int*   mask  = (const int*)  d_in[3];   // [B,SQ,SK]
    const float* Wq    = (const float*)d_in[4];   // [D,S]
    const float* bq    = (const float*)d_in[5];   // [S]
    const float* Wk    = (const float*)d_in[6];   // [D,S]
    const float* bk    = (const float*)d_in[7];   // [S]
    const float* W1    = (const float*)d_in[8];   // [2S,S]
    const float* b1    = (const float*)d_in[9];   // [S]
    const float* W2    = (const float*)d_in[10];  // [S,1]
    const float* b2    = (const float*)d_in[11];  // [1]

    float* out_f  = (float*)d_out;                         // [B,SQ,D]
    float* attn_f = (float*)d_out + (long)Bz * SQ * Dd;    // [B,SQ,SK]

    // workspace layout (256B aligned slabs)
    char* w = (char*)d_ws;
    auto alloc = [&](size_t bytes) -> void* {
        void* p = (void*)w;
        w += (bytes + 255) & ~(size_t)255;
        return p;
    };
    __bf16* q_b    = (__bf16*)alloc((size_t)Bz * SQ * Dd * 2);   // [2048,512]
    __bf16* k_b    = (__bf16*)alloc((size_t)Bz * SK * Dd * 2);
    __bf16* WqT    = (__bf16*)alloc((size_t)Ss * Dd * 2);        // [128,512]
    __bf16* WkT    = (__bf16*)alloc((size_t)Ss * Dd * 2);
    __bf16* W1tT   = (__bf16*)alloc((size_t)Ss * Ss * 2);        // [128,128]
    __bf16* W1bT   = (__bf16*)alloc((size_t)Ss * Ss * 2);
    __bf16* pq_b   = (__bf16*)alloc((size_t)Bz * SQ * Ss * 2);   // [2048,128]
    __bf16* pk_b   = (__bf16*)alloc((size_t)Bz * SK * Ss * 2);
    float*  qh     = (float*) alloc((size_t)Bz * SQ * Ss * 4);   // [2048,128]
    float*  kh     = (float*) alloc((size_t)Bz * SK * Ss * 4);
    __bf16* vT     = (__bf16*)alloc((size_t)Bz * Dd * SK * 2);   // [B,D,SK]
    __bf16* attn_b = (__bf16*)alloc((size_t)Bz * SQ * SK * 2);   // [B,SQ,SK]

    const long NQ = (long)Bz * SQ * Dd;   // 1,048,576

    // 1) converts / transposes
    cvt_f32_bf16<<<dim3(2048), dim3(256), 0, stream>>>(query, q_b, NQ);
    cvt_f32_bf16<<<dim3(2048), dim3(256), 0, stream>>>(key,   k_b, NQ);
    transpose_cvt<<<dim3(256, 1), dim3(256), 0, stream>>>(Wq, WqT, Dd, Ss);
    transpose_cvt<<<dim3(256, 1), dim3(256), 0, stream>>>(Wk, WkT, Dd, Ss);
    transpose_cvt<<<dim3(64,  1), dim3(256), 0, stream>>>(W1,            W1tT, Ss, Ss);
    transpose_cvt<<<dim3(64,  1), dim3(256), 0, stream>>>(W1 + Ss * Ss,  W1bT, Ss, Ss);
    transpose_cvt<<<dim3(512, Bz), dim3(256), 0, stream>>>(value, vT, SK, Dd);

    // 2) pq = q @ Wq + bq   (bf16 out)   M=2048 N=128 K=512, 16x64 tiles
    gemm_bf16_wmma<<<dim3(Bz * SQ / 16, Ss / 64, 1), dim3(32), 0, stream>>>(
        q_b, WqT, bq, nullptr, pq_b, Bz * SQ, Ss, Dd, 0, 0, 0);
    gemm_bf16_wmma<<<dim3(Bz * SK / 16, Ss / 64, 1), dim3(32), 0, stream>>>(
        k_b, WkT, bk, nullptr, pk_b, Bz * SK, Ss, Dd, 0, 0, 0);

    // 3) qh = pq @ W1[:S],  kh = pk @ W1[S:]   (f32 out)  M=2048 N=128 K=128
    gemm_bf16_wmma<<<dim3(Bz * SQ / 16, Ss / 64, 1), dim3(32), 0, stream>>>(
        pq_b, W1tT, nullptr, qh, nullptr, Bz * SQ, Ss, Ss, 0, 0, 0);
    gemm_bf16_wmma<<<dim3(Bz * SK / 16, Ss / 64, 1), dim3(32), 0, stream>>>(
        pk_b, W1bT, nullptr, kh, nullptr, Bz * SK, Ss, Ss, 0, 0, 0);

    // 4) fused scores + mask + softmax
    scores_softmax<<<dim3(SQ, Bz), dim3(256), 0, stream>>>(
        qh, kh, b1, W2, b2, mask, attn_f, attn_b);

    // 5) out = attn @ value   (f32 out)  per batch: M=512 N=512 K=512, 16x64 tiles
    gemm_bf16_wmma<<<dim3(SQ / 16, Dd / 64, Bz), dim3(32), 0, stream>>>(
        attn_b, vT, nullptr, out_f, nullptr, SQ, Dd, SK,
        (long)SQ * SK, (long)Dd * SK, (long)SQ * Dd);
}
